// PoolAggregator_33698313404802
// MI455X (gfx1250) — matrix operand order repaired and checked by reading, not compiled, not run on hardware
//
#include <hip/hip_runtime.h>

typedef __attribute__((ext_vector_type(16))) __bf16        v16bf;
typedef __attribute__((ext_vector_type(8)))  float         v8f;
typedef __attribute__((ext_vector_type(4)))  float         f32x4;
typedef __attribute__((ext_vector_type(4)))  unsigned int  u32x4;
typedef __attribute__((ext_vector_type(2)))  unsigned int  u32x2;

#define D_DIM 512
#define S_NB  32
#define XSTR  520   // LDS row stride in bf16 elements (512 + 8 pad -> bank-spread)

__device__ __forceinline__ unsigned short f2bf(float f) {
  union { float f; unsigned u; } c; c.f = f;
  unsigned u = c.u;
  unsigned r = u + 0x7FFFu + ((u >> 16) & 1u);   // round-to-nearest-even
  return (unsigned short)(r >> 16);
}

// One-time (per launch) conversion of W [512,512] f32 -> bf16 into workspace.
__global__ void __launch_bounds__(256)
cvt_w_bf16(const float* __restrict__ W, unsigned short* __restrict__ Wbf) {
  const int i = (blockIdx.x * 256 + threadIdx.x) * 4;
  f32x4 f = *(const f32x4*)(W + i);
  u32x2 p;
  p.x = (unsigned)f2bf(f.x) | ((unsigned)f2bf(f.y) << 16);
  p.y = (unsigned)f2bf(f.z) | ((unsigned)f2bf(f.w) << 16);
  *(u32x2*)(Wbf + i) = p;
}

// One workgroup per batch node: gather 32 neighbor rows -> LDS (bf16),
// GEMM [32 x 512] x [512 x 512]^T via v_wmma_f32_16x16x32_bf16,
// then relu + mean over the 32 rows.
__global__ void __launch_bounds__(256)
pool_agg_kernel(const int* __restrict__ nidx,
                const float* __restrict__ feat,
                const unsigned short* __restrict__ Wbf,
                const float* __restrict__ bias,
                float* __restrict__ out) {
  __shared__ unsigned short Xlds[S_NB * XSTR];

  const int b = blockIdx.x;
  const int t = threadIdx.x;

  // ---- gather + f32->bf16 convert into LDS: 32 rows x 512 cols ----
  {
    const int r     = t >> 3;            // neighbor slot 0..31 (8 threads/row)
    const int cbase = (t & 7) * 64;      // 64 floats per thread
    const long rowoff = (long)nidx[b * S_NB + r] * D_DIM;
    const float* src = feat + rowoff + cbase;
    unsigned short* dst = Xlds + r * XSTR + cbase;
#pragma unroll
    for (int i = 0; i < 16; ++i) {
      f32x4 f = *(const f32x4*)(src + i * 4);
      u32x2 p;
      p.x = (unsigned)f2bf(f.x) | ((unsigned)f2bf(f.y) << 16);
      p.y = (unsigned)f2bf(f.z) | ((unsigned)f2bf(f.w) << 16);
      *(u32x2*)(dst + i * 4) = p;
    }
  }
  __syncthreads();

  const int lane  = t & 31;
  const int wave  = t >> 5;
  const int half  = lane >> 4;   // K-half selector per ISA fragment layout
  const int m16   = lane & 15;
  const int nbase = wave * 64;   // 8 waves x 64 columns = 512

  v8f acc[2][4] = {};            // M-tiles {0,1} x N-tiles {0..3}

  for (int kk = 0; kk < D_DIM; kk += 32) {
    // A fragments (16x32 bf16): lane holds row m16, 16 contiguous K at 16*half
    v16bf amat[2];
#pragma unroll
    for (int mt = 0; mt < 2; ++mt) {
      const unsigned short* rp = Xlds + (mt * 16 + m16) * XSTR + kk + 16 * half;
      union { u32x4 q[2]; v16bf v; } u;
      u.q[0] = *(const u32x4*)(rp);        // K = kk + 16*half .. +7
      u.q[1] = *(const u32x4*)(rp + 8);    // K = kk + 16*half + 8 .. +7
      amat[mt] = u.v;
    }
    // B fragments (32x16 bf16): lane holds column n = row of W, 16 contiguous K
    v16bf bmat[4];
#pragma unroll
    for (int nt = 0; nt < 4; ++nt) {
      const unsigned short* wp =
          Wbf + (size_t)(nbase + nt * 16 + m16) * D_DIM + kk + 16 * half;
      bmat[nt] = *(const v16bf*)wp;
    }
#pragma unroll
    for (int mt = 0; mt < 2; ++mt)
#pragma unroll
      for (int nt = 0; nt < 4; ++nt)
        acc[mt][nt] = __builtin_amdgcn_wmma_f32_16x16x32_bf16(
            false, amat[mt], false, bmat[nt], (short)0, acc[mt][nt],
            false, false);
  }

  // ---- bias + relu + mean over the 32 neighbor rows ----
  const float inv = 1.0f / 32.0f;
#pragma unroll
  for (int nt = 0; nt < 4; ++nt) {
    const int n = nbase + nt * 16 + m16;
    const float bb = bias[n];
    float s = 0.0f;
#pragma unroll
    for (int mt = 0; mt < 2; ++mt)
#pragma unroll
      for (int r = 0; r < 8; ++r) {
        float v = acc[mt][nt][r] + bb;   // C layout: M = r + 8*half, N = m16
        s += fmaxf(v, 0.0f);
      }
    s += __shfl_xor(s, 16, 32);          // combine the two M-halves per column
    if (lane < 16) out[(size_t)b * D_DIM + n] = s * inv;
  }
}

extern "C" void kernel_launch(void* const* d_in, const int* in_sizes, int n_in,
                              void* d_out, int out_size, void* d_ws, size_t ws_size,
                              hipStream_t stream) {
  const int*   nidx = (const int*)d_in[0];     // [10000, 32] int32
  const float* feat = (const float*)d_in[1];   // [100000, 512] f32
  const float* W    = (const float*)d_in[2];   // [512, 512] f32
  const float* bias = (const float*)d_in[3];   // [512] f32
  float* out = (float*)d_out;                  // [10000, 512] f32
  unsigned short* Wbf = (unsigned short*)d_ws; // 512*512*2 = 512 KB scratch

  (void)in_sizes; (void)n_in; (void)out_size; (void)ws_size;

  // W f32 -> bf16 (262144 elems, 4 per thread)
  cvt_w_bf16<<<256, 256, 0, stream>>>(W, Wbf);
  // One workgroup per batch node
  pool_agg_kernel<<<10000, 256, 0, stream>>>(nidx, feat, Wbf, bias, out);
}
